// SpherConvLSTM_EncoderCell_68813966017200
// MI455X (gfx1250) — matrix operand order repaired and checked by reading, not compiled, not run on hardware
//
#include <hip/hip_runtime.h>
#include <hip/hip_bf16.h>

typedef __attribute__((ext_vector_type(16))) _Float16 v16h;
typedef __attribute__((ext_vector_type(8)))  float    v8f;

// ---- CDNA5 WMMA 16x16x32 f16 fragment index helpers (wave32, contiguous k order used by this file, not the ISA table order) ----
// A (16x32, M x K): lanes 0-15 -> M=lane, K in {0..15}; lanes 16-31 -> K in {16..31}
__device__ __forceinline__ int wmma_a_k(int e, int lane) {
  int kb = ((lane >> 4) & 1) << 4;               // 0 or 16
  return kb + e;
}
// B (32x16, K x N): N = lane&15 ; K = (lane>=16 ? 16 : 0) + e  -> 16 CONSECUTIVE k per lane
__device__ __forceinline__ int wmma_b_k(int e, int lane) {
  return (((lane >> 4) & 1) << 4) + e;
}
__device__ __forceinline__ float rcp_(float v)      { return __builtin_amdgcn_rcpf(v); }
__device__ __forceinline__ float sigmoidf_(float v) { return rcp_(1.0f + __expf(-v)); }
__device__ __forceinline__ float tanhf_(float v)    { return 1.0f - 2.0f * rcp_(__expf(2.0f * v) + 1.0f); }

// packed-f16 weight workspace (halves): Wl[128x320] Wu[128x288] Wd[32x1152] Wqw[256x128] Wqh[256x128]
#define WL16_N  (128 * 320)
#define WU16_N  (128 * 288)
#define WD16_N  (32 * 1152)
#define WQ16_N  (256 * 128)

// ============================================================================
// Kernel P: pack weights into WMMA A-fragment order:
//   dst[((mt*CH + ch)*32 + lane)*16 + e] = src[(mt*16 + lane%16)*Korig + ch*32 + a_k(e,lane)]
// Each lane's fragment becomes one contiguous 32-byte block.
// ============================================================================
__device__ __forceinline__ void pack_frag(const float* __restrict__ src, _Float16* __restrict__ dst,
                                          int M, int K, int Korig, int tid, int stride) {
  const int CH = K / 32;
  const int total = M * K;
  for (int i = tid; i < total; i += stride) {
    int e = i & 15;
    int lane = (i >> 4) & 31;
    int j = i >> 9;                       // mt*CH + ch
    int ch = j % CH, mt = j / CH;
    int k = ch * 32 + wmma_a_k(e, lane);
    int mrow = mt * 16 + (lane & 15);
    dst[i] = (_Float16)((k < Korig) ? src[mrow * Korig + k] : 0.f);
  }
}

__global__ void k_prep_weights(const float* __restrict__ Wl, const float* __restrict__ Wu,
                               const float* __restrict__ Wd, const float* __restrict__ Wqw,
                               const float* __restrict__ Wqh, _Float16* __restrict__ dst) {
  const int tid = blockIdx.x * 256 + threadIdx.x;
  const int stride = gridDim.x * 256;
  _Float16* wl = dst;
  _Float16* wu = wl + WL16_N;
  _Float16* wd = wu + WU16_N;
  _Float16* wqw = wd + WD16_N;
  _Float16* wqh = wqw + WQ16_N;
  pack_frag(Wl,  wl,  128, 320, 315, tid, stride);
  pack_frag(Wu,  wu,  128, 288, 288, tid, stride);
  pack_frag(Wd,  wd,  32, 1152, 1152, tid, stride);
  pack_frag(Wqw, wqw, 256, 128, 128, tid, stride);
  pack_frag(Wqh, wqh, 256, 128, 128, tid, stride);
}

// ============================================================================
// Kernel A: ConvLSTM: cc = conv3x3(concat(x,h0), W_lstm)+b ; gates -> h, c
// Grid (5, 240, 8); block 256. Per WG: 64-px row segment, all 128 cc channels.
// LDS: raw f16 tile -> pixel-major im2col tile [64][320] -> f32 cc buffer.
// ============================================================================
__global__ void k_conv_lstm(const float* __restrict__ xin, const float* __restrict__ h0,
                            const float* __restrict__ c0, const _Float16* __restrict__ Wl16,
                            const float* __restrict__ bl, float* __restrict__ hOut,
                            float* __restrict__ cOut) {
  extern __shared__ __align__(32) char smemc[];
  _Float16* rawS  = (_Float16*)smemc;            // 35*3*66 = 6930 h (13860 B)
  _Float16* tileI = (_Float16*)(smemc + 13888);  // [64 px][320 k] = 40960 B
  float* ccS = (float*)(smemc + 54848);          // 128*64 floats (32768 B)
  const int seg = blockIdx.x, y = blockIdx.y, b = blockIdx.z;
  const int x0 = seg * 64;
  const int tid = threadIdx.x, lane = tid & 31, wave = tid >> 5;
  const int m = lane & 15, hi = (lane >> 4) & 1;

  __builtin_prefetch(Wl16 + (wave * 10) * 512 + lane * 16, 0, 0);  // global_prefetch_b8

  for (int idx = tid; idx < 35 * 3 * 66; idx += 256) {
    int cin = idx / (3 * 66), rem = idx % (3 * 66);
    int ry = rem / 66, col = rem % 66;
    int gy = y + ry - 1, gx = x0 + col - 1;
    float v = 0.f;
    if (gy >= 0 && gy < 240 && gx >= 0 && gx < 320)
      v = (cin < 3) ? xin[((long)(b * 3 + cin) * 240 + gy) * 320 + gx]
                    : h0[((long)(b * 32 + (cin - 3)) * 240 + gy) * 320 + gx];
    rawS[idx] = (_Float16)v;
  }
  __syncthreads();

  // im2col expand: decode k once per thread, stream pixels
  for (int k = tid; k < 320; k += 256) {
    if (k < 315) {
      int cin = k / 9, r2 = k % 9, ky = r2 / 3, kx = r2 % 3;
      int rb = (cin * 3 + ky) * 66 + kx;
      for (int px = 0; px < 64; ++px) tileI[px * 320 + k] = rawS[rb + px];
    } else {
      for (int px = 0; px < 64; ++px) tileI[px * 320 + k] = (_Float16)0.f;
    }
  }
  __syncthreads();

  {
    v8f acc0 = {}, acc1 = {}, acc2 = {}, acc3 = {};
    const v16h* Ap = (const v16h*)Wl16;
    const v16h* b0 = (const v16h*)(tileI + (0 * 16 + m) * 320);
    const v16h* b1 = (const v16h*)(tileI + (1 * 16 + m) * 320);
    const v16h* b2 = (const v16h*)(tileI + (2 * 16 + m) * 320);
    const v16h* b3 = (const v16h*)(tileI + (3 * 16 + m) * 320);
    for (int ch = 0; ch < 10; ++ch) {
      v16h a = Ap[(wave * 10 + ch) * 32 + lane];
      int bi = ch * 2 + hi;
      acc0 = __builtin_amdgcn_wmma_f32_16x16x32_f16(false, a, false, b0[bi], (short)0, acc0, false, false);
      acc1 = __builtin_amdgcn_wmma_f32_16x16x32_f16(false, a, false, b1[bi], (short)0, acc1, false, false);
      acc2 = __builtin_amdgcn_wmma_f32_16x16x32_f16(false, a, false, b2[bi], (short)0, acc2, false, false);
      acc3 = __builtin_amdgcn_wmma_f32_16x16x32_f16(false, a, false, b3[bi], (short)0, acc3, false, false);
    }
#pragma unroll
    for (int r = 0; r < 8; ++r) {
      int o = wave * 16 + r + hi * 8;
      float bb = bl[o];
      ccS[o * 64 +  0 + m] = acc0[r] + bb;
      ccS[o * 64 + 16 + m] = acc1[r] + bb;
      ccS[o * 64 + 32 + m] = acc2[r] + bb;
      ccS[o * 64 + 48 + m] = acc3[r] + bb;
    }
  }
  __syncthreads();

  for (int idx = tid; idx < 32 * 64; idx += 256) {
    int ch = idx >> 6, px = idx & 63;
    float iv = sigmoidf_(ccS[ch * 64 + px]);
    float fv = sigmoidf_(ccS[(ch + 32) * 64 + px]);
    float ov = sigmoidf_(ccS[(ch + 64) * 64 + px]);
    float gv = tanhf_(ccS[(ch + 96) * 64 + px]);
    long off = ((long)(b * 32 + ch) * 240 + y) * 320 + x0 + px;
    float cv = fv * c0[off] + iv * gv;
    hOut[off] = ov * tanhf_(cv);
    cOut[off] = cv;
  }
}

// ============================================================================
// Kernel B: conv3x3(h, W_u)+b -> 4x4 maxpool -> + prev_memory -> xin_w (ws)
// Grid (5, 60, 8); block 256. Per WG: 16 pooled px (64 conv cols x 4 rows).
// LDS: raw [32][6][66] f16 (25344 B) | im2col [256 cp][288] (147456 B) | conv f32 (131072 B)
// ============================================================================
__global__ void k_conv_u_pool(const float* __restrict__ h, const _Float16* __restrict__ Wu16,
                              const float* __restrict__ bu, const float* __restrict__ pmem,
                              float* __restrict__ xinW) {
  extern __shared__ __align__(32) char smemc[];
  _Float16* rawS  = (_Float16*)smemc;
  _Float16* tileI = (_Float16*)(smemc + 25344);
  float* convS = (float*)(smemc + 172800);
  const int seg = blockIdx.x, py = blockIdx.y, b = blockIdx.z;
  const int x0 = seg * 64, y0 = py * 4;
  const int tid = threadIdx.x, lane = tid & 31, wave = tid >> 5;
  const int m = lane & 15, hi = (lane >> 4) & 1;

  __builtin_prefetch(Wu16 + (wave * 9) * 512 + lane * 16, 0, 0);

  for (int idx = tid; idx < 32 * 6 * 66; idx += 256) {
    int cin = idx / (6 * 66), rem = idx % (6 * 66);
    int ry = rem / 66, col = rem % 66;
    int gy = y0 + ry - 1, gx = x0 + col - 1;
    float v = 0.f;
    if (gy >= 0 && gy < 240 && gx >= 0 && gx < 320)
      v = h[((long)(b * 32 + cin) * 240 + gy) * 320 + gx];
    rawS[idx] = (_Float16)v;
  }
  __syncthreads();

  for (int k = tid; k < 288; k += 256) {
    int cin = k / 9, r2 = k % 9, ky = r2 / 3, kx = r2 % 3;
    int rb = cin * 396 + ky * 66 + kx;
    for (int cp = 0; cp < 256; ++cp) {       // cp = cr*64 + px
      int cr = cp >> 6, px = cp & 63;
      tileI[cp * 288 + k] = rawS[rb + cr * 66 + px];
    }
  }
  __syncthreads();

  for (int cr = 0; cr < 4; ++cr) {
    v8f acc0 = {}, acc1 = {}, acc2 = {}, acc3 = {};
    const v16h* Ap = (const v16h*)Wu16;
    const v16h* b0 = (const v16h*)(tileI + (cr * 64 +  0 + m) * 288);
    const v16h* b1 = (const v16h*)(tileI + (cr * 64 + 16 + m) * 288);
    const v16h* b2 = (const v16h*)(tileI + (cr * 64 + 32 + m) * 288);
    const v16h* b3 = (const v16h*)(tileI + (cr * 64 + 48 + m) * 288);
    for (int ch = 0; ch < 9; ++ch) {
      v16h a = Ap[(wave * 9 + ch) * 32 + lane];
      int bi = ch * 2 + hi;
      acc0 = __builtin_amdgcn_wmma_f32_16x16x32_f16(false, a, false, b0[bi], (short)0, acc0, false, false);
      acc1 = __builtin_amdgcn_wmma_f32_16x16x32_f16(false, a, false, b1[bi], (short)0, acc1, false, false);
      acc2 = __builtin_amdgcn_wmma_f32_16x16x32_f16(false, a, false, b2[bi], (short)0, acc2, false, false);
      acc3 = __builtin_amdgcn_wmma_f32_16x16x32_f16(false, a, false, b3[bi], (short)0, acc3, false, false);
    }
#pragma unroll
    for (int r = 0; r < 8; ++r) {
      int o = wave * 16 + r + hi * 8;
      float bb = bu[o];
      convS[(o * 4 + cr) * 64 +  0 + m] = acc0[r] + bb;
      convS[(o * 4 + cr) * 64 + 16 + m] = acc1[r] + bb;
      convS[(o * 4 + cr) * 64 + 32 + m] = acc2[r] + bb;
      convS[(o * 4 + cr) * 64 + 48 + m] = acc3[r] + bb;
    }
  }
  __syncthreads();

  for (int idx = tid; idx < 128 * 16; idx += 256) {
    int ch = idx >> 4, pp = idx & 15;
    float mx = -3.0e38f;
    for (int cr = 0; cr < 4; ++cr)
      for (int cc = 0; cc < 4; ++cc)
        mx = fmaxf(mx, convS[(ch * 4 + cr) * 64 + pp * 4 + cc]);
    long poff = ((long)(b * 128 + ch) * 60 + py) * 80 + seg * 16 + pp;
    xinW[poff] = mx + pmem[poff];
  }
}

// ============================================================================
// Kernel C: axial attention, one sequence per WG (template: compile-time dims).
// ============================================================================
template <int L, int LP, int RelW>
__global__ void k_axial_attn(const float* __restrict__ xsrc, float* __restrict__ outp,
                             const _Float16* __restrict__ Wq16, const float* __restrict__ gk,
                             const float* __restrict__ bkq, const float* __restrict__ gl,
                             const float* __restrict__ blg, const float* __restrict__ rel,
                             int S2, int sL, int sR, float inScale) {
  extern __shared__ __align__(32) char smemc[];
  float* kqvS = (float*)smemc;                   // 256*LP
  float* xinS = kqvS + 256 * LP;                 // 128*LP
  float* logS = xinS;                            // alias after kqv is done
  constexpr int NT = LP / 16;
  constexpr int KC = LP / 32;
  const int seq = blockIdx.x;
  const int b = seq / S2, rr = seq % S2;
  const int tid = threadIdx.x, lane = tid & 31, wave = tid >> 5;
  const int m = lane & 15, hi = (lane >> 4) & 1;
  const long base = (long)b * 128 * 4800;

  __builtin_prefetch(Wq16 + tid * 128, 0, 0);

  for (int idx = tid; idx < 128 * LP; idx += 256) {
    int c = idx / LP, l = idx % LP;
    xinS[idx] = (l < L) ? inScale * xsrc[base + (long)c * 4800 + l * sL + rr * sR] : 0.f;
  }
  __syncthreads();

  // kqv = g * (Wkqv @ xin) + b ; M=256, N=LP, K=128 (packed A fragments)
  for (int job = wave; job < 16 * NT; job += 8) {
    int mt = job / NT, nt = job % NT, o0 = mt * 16;
    v8f acc = {};
    const v16h* Ap = (const v16h*)Wq16;
    for (int ch = 0; ch < 4; ++ch) {
      v16h a = Ap[(mt * 4 + ch) * 32 + lane];
      v16h bf;
      const int kc = ch * 32;
#pragma unroll
      for (int e = 0; e < 16; ++e)
        bf[e] = (_Float16)xinS[(kc + wmma_b_k(e, lane)) * LP + nt * 16 + m];
      acc = __builtin_amdgcn_wmma_f32_16x16x32_f16(false, a, false, bf, (short)0, acc, false, false);
    }
#pragma unroll
    for (int r = 0; r < 8; ++r) {
      int o = o0 + r + hi * 8;
      kqvS[o * LP + nt * 16 + m] = gk[o] * acc[r] + bkq[o];
    }
  }
  __syncthreads();

  // logits = g0 * q^T k + b0 ; q=kqv[64:128], k=kqv[0:64] ; K=64
  for (int job = wave; job < NT * NT; job += 8) {
    int xt = job / NT, yt = job % NT;
    v8f acc = {};
    for (int ch = 0; ch < 2; ++ch) {
      const int kc = ch * 32;
      v16h a, bf;
#pragma unroll
      for (int e = 0; e < 16; ++e)
        a[e] = (_Float16)kqvS[(64 + kc + wmma_a_k(e, lane)) * LP + xt * 16 + m];
#pragma unroll
      for (int e = 0; e < 16; ++e)
        bf[e] = (_Float16)kqvS[(kc + wmma_b_k(e, lane)) * LP + yt * 16 + m];
      acc = __builtin_amdgcn_wmma_f32_16x16x32_f16(false, a, false, bf, (short)0, acc, false, false);
    }
#pragma unroll
    for (int r = 0; r < 8; ++r)
      logS[(xt * 16 + r + hi * 8) * LP + yt * 16 + m] = gl[0] * acc[r] + blg[0];
  }
  __syncthreads();

  // += g1*qr + b1 + g2*kr + b2 (rel table is L2-hot)
  for (int idx = tid; idx < L * L; idx += 256) {
    int xx = idx / L, yy = idx % L;
    int dq = yy - xx + L - 1, dk = xx - yy + L - 1;
    float qr = 0.f, kr = 0.f;
#pragma unroll 4
    for (int d = 0; d < 64; ++d) {
      qr += kqvS[(64 + d) * LP + xx] * rel[d * RelW + dq];
      kr += kqvS[d * LP + yy]        * rel[(64 + d) * RelW + dk];
    }
    logS[xx * LP + yy] += gl[1] * qr + blg[1] + gl[2] * kr + blg[2];
  }
  __syncthreads();

  // softmax rows; zero pad cols so attn GEMM needs no masking
  for (int xx = tid; xx < L; xx += 256) {
    float* row = logS + xx * LP;
    float mx = -3.0e38f;
    for (int yy = 0; yy < L; ++yy) mx = fmaxf(mx, row[yy]);
    float s = 0.f;
    for (int yy = 0; yy < L; ++yy) { float e = __expf(row[yy] - mx); row[yy] = e; s += e; }
    float inv = rcp_(s);
    for (int yy = 0; yy < L; ++yy) row[yy] *= inv;
    for (int yy = L; yy < LP; ++yy) row[yy] = 0.f;
  }
  __syncthreads();

  // attn = v @ wts^T (WMMA) + sum_y wts[x,y]*rel_v[d, y-x+L-1] (scalar)
  for (int job = wave; job < 8 * NT; job += 8) {
    int dt = job / NT, xt = job % NT, d0 = dt * 16;
    v8f acc = {};
    for (int ch = 0; ch < KC; ++ch) {
      const int kc = ch * 32;
      v16h a, bf;
#pragma unroll
      for (int e = 0; e < 16; ++e)
        a[e] = (_Float16)kqvS[(128 + d0 + m) * LP + kc + wmma_a_k(e, lane)];
#pragma unroll
      for (int e = 0; e < 16; ++e)
        bf[e] = (_Float16)logS[(xt * 16 + m) * LP + kc + wmma_b_k(e, lane)];
      acc = __builtin_amdgcn_wmma_f32_16x16x32_f16(false, a, false, bf, (short)0, acc, false, false);
    }
#pragma unroll
    for (int r = 0; r < 8; ++r) {
      int d = d0 + r + hi * 8;
      int xx = xt * 16 + m;
      if (xx < L) {
        float a2 = 0.f;
        const float* rv = rel + (long)(128 + d) * RelW + (L - 1 - xx);
#pragma unroll 4
        for (int yy = 0; yy < L; ++yy) a2 += logS[xx * LP + yy] * rv[yy];
        outp[base + (long)d * 4800 + xx * sL + rr * sR] = acc[r] + a2;
      }
    }
  }
}

// ============================================================================
// Kernel E: conv3x3(mem, W_d)+b -> 2x2 nearest upsample -> d_out[out]
// Grid (60, 8); block 256. Per WG: one 80-px row, 32 out channels.
// LDS: raw [128][3][82] f16 (62976 B) | im2col [80 px][1152] (184320 B)
// ============================================================================
__global__ void k_conv_d_up(const float* __restrict__ memIn, const _Float16* __restrict__ Wd16,
                            const float* __restrict__ bd, float* __restrict__ outF) {
  extern __shared__ __align__(32) char smemc[];
  _Float16* rawS  = (_Float16*)smemc;
  _Float16* tileI = (_Float16*)(smemc + 62976);
  const int y = blockIdx.x, b = blockIdx.y;
  const int tid = threadIdx.x, lane = tid & 31, wave = tid >> 5;
  const int m = lane & 15, hi = (lane >> 4) & 1;

  __builtin_prefetch(Wd16 + tid * 144, 0, 0);

  for (int idx = tid; idx < 128 * 3 * 82; idx += 256) {
    int cin = idx / (3 * 82), rem = idx % (3 * 82);
    int ry = rem / 82, col = rem % 82;
    int gy = y + ry - 1, gx = col - 1;
    float v = 0.f;
    if (gy >= 0 && gy < 60 && gx >= 0 && gx < 80)
      v = memIn[((long)(b * 128 + cin) * 60 + gy) * 80 + gx];
    rawS[idx] = (_Float16)v;
  }
  __syncthreads();

  for (int k = tid; k < 1152; k += 256) {
    int cin = k / 9, r2 = k % 9, ky = r2 / 3, kx = r2 % 3;
    int rb = cin * 246 + ky * 82 + kx;
    for (int px = 0; px < 80; ++px) tileI[px * 1152 + k] = rawS[rb + px];
  }
  __syncthreads();

  for (int job = wave; job < 10; job += 8) {     // 2 M-tiles x 5 N-subtiles
    int mt = job / 5, sub = job % 5, o0 = mt * 16;
    v8f acc = {};
    const v16h* Ap = (const v16h*)Wd16;
    const v16h* brow = (const v16h*)(tileI + (sub * 16 + m) * 1152);
    for (int ch = 0; ch < 36; ++ch) {
      v16h a = Ap[(mt * 36 + ch) * 32 + lane];
      acc = __builtin_amdgcn_wmma_f32_16x16x32_f16(false, a, false, brow[ch * 2 + hi], (short)0, acc, false, false);
    }
#pragma unroll
    for (int r = 0; r < 8; ++r) {
      int o = o0 + r + hi * 8;
      int xx = sub * 16 + m;
      float v = acc[r] + bd[o];
      long bo = ((long)(b * 32 + o) * 120 + 2 * y) * 160 + 2 * xx;
      outF[bo] = v; outF[bo + 1] = v; outF[bo + 160] = v; outF[bo + 161] = v;
    }
  }
}

// ============================================================================
extern "C" void kernel_launch(void* const* d_in, const int* in_sizes, int n_in,
                              void* d_out, int out_size, void* d_ws, size_t ws_size,
                              hipStream_t stream) {
  (void)in_sizes; (void)n_in; (void)out_size; (void)ws_size;
  const float* x       = (const float*)d_in[0];
  const float* h0      = (const float*)d_in[1];
  const float* c0      = (const float*)d_in[2];
  const float* pmem    = (const float*)d_in[3];
  const float* W_lstm  = (const float*)d_in[4];
  const float* b_lstm  = (const float*)d_in[5];
  const float* W_u     = (const float*)d_in[6];
  const float* b_u     = (const float*)d_in[7];
  const float* W_d     = (const float*)d_in[8];
  const float* b_d     = (const float*)d_in[9];
  const float* Wkqv_w  = (const float*)d_in[10];
  const float* g_kqv_w = (const float*)d_in[11];
  const float* b_kqv_w = (const float*)d_in[12];
  const float* g_log_w = (const float*)d_in[13];
  const float* b_log_w = (const float*)d_in[14];
  const float* rel_w   = (const float*)d_in[15];
  const float* Wkqv_h  = (const float*)d_in[16];
  const float* g_kqv_h = (const float*)d_in[17];
  const float* b_kqv_h = (const float*)d_in[18];
  const float* g_log_h = (const float*)d_in[19];
  const float* b_log_h = (const float*)d_in[20];
  const float* rel_h   = (const float*)d_in[21];

  float* out   = (float*)d_out;                  // (8,32,120,160)
  float* hOut  = out + 1228800;                  // (8,32,240,320)
  float* cOut  = out + 20889600;                 // (8,32,240,320)
  float* mem   = out + 40550400;                 // (8,128,60,80)
  float* ws    = (float*)d_ws;
  float* xinW  = ws;                             // (8,128,60,80) pooled+prev_memory
  float* wOut  = ws + 4915200;                   // (8,128,60,80) width-attn output
  _Float16* w16 = (_Float16*)(ws + 9830400);     // packed f16 weights (~360 KB)
  _Float16* Wl16  = w16;
  _Float16* Wu16  = Wl16 + WL16_N;
  _Float16* Wd16  = Wu16 + WU16_N;
  _Float16* Wqw16 = Wd16 + WD16_N;
  _Float16* Wqh16 = Wqw16 + WQ16_N;

  // 0) weight pack -> WMMA fragment order, f16 (L2-resident for all later kernels)
  k_prep_weights<<<64, 256, 0, stream>>>(W_lstm, W_u, W_d, Wkqv_w, Wkqv_h, w16);
  // 1) ConvLSTM -> h, c
  k_conv_lstm<<<dim3(5, 240, 8), 256, 87616, stream>>>(
      x, h0, c0, Wl16, b_lstm, hOut, cOut);
  // 2) conv_u + 4x4 maxpool + prev_memory
  k_conv_u_pool<<<dim3(5, 60, 8), 256, 303872, stream>>>(
      hOut, Wu16, b_u, pmem, xinW);
  // 3) width axial attention: L=80, LP=96
  k_axial_attn<80, 96, 159><<<480, 256, 384 * 96 * 4, stream>>>(
      xinW, wOut, Wqw16, g_kqv_w, b_kqv_w, g_log_w, b_log_w, rel_w,
      60, 1, 80, 1.0f);
  // 4) height axial attention: input = out+mem = 2*wOut; L=60, LP=64; writes mem slot
  k_axial_attn<60, 64, 119><<<640, 256, 384 * 64 * 4, stream>>>(
      wOut, mem, Wqh16, g_kqv_h, b_kqv_h, g_log_h, b_log_h, rel_h,
      80, 80, 1, 2.0f);
  // 5) conv_d + 2x upsample
  k_conv_d_up<<<dim3(60, 8), 256, 62976 + 184320, stream>>>(
      mem, Wd16, b_d, out);
}
